// EdgeClassificationGNN_61615600828943
// MI455X (gfx1250) — compile-verified, matrix operand order checked by reading, not run on hardware
//
#include <hip/hip_runtime.h>

// ---------------------------------------------------------------------------
// CDNA5 (gfx1250) fused GNN implementation.
// Matrix work: v_wmma_f32_16x16x32_bf16 (wave32 WMMA, f32 accumulate) with
// fragment-native (pre-swizzled) operand layouts: one 32-byte vector load per
// fragment per lane, dual-accumulator register blocking on wide layers.
// ---------------------------------------------------------------------------

typedef __attribute__((ext_vector_type(16))) __bf16 v16bf;
typedef __attribute__((ext_vector_type(8)))  float  v8f;

#define TPB 256
#define MAX_KPAD   608              // gat layer1: 579 -> 608
#define MAX_KTILES (MAX_KPAD / 32)  // 19
#define MAX_NOUT   512

__device__ __forceinline__ v8f wmma_bf16(v16bf a, v16bf b, v8f c) {
  return __builtin_amdgcn_wmma_f32_16x16x32_bf16(
      false, a, false, b, (short)0, c, false, false);
}

// ---------------- generic fused MLP layer ----------------------------------
// Y[M x Nout] = act( X[M x Kin] @ W + b ), act: 0 = none, 1 = tanh(LayerNorm)
// One workgroup = 16 rows; 8 waves cover 16-col N tiles; K loop uses WMMA.
//
// LDS X staging is written in WMMA A-fragment order:
//   sX[kt*512 + fl*16 + j], fl = m + 16*((ko>>3)&1), j = 8*(ko>>4) + (ko&7)
// (ISA 7.12.2: lanes 0-15 hold K 0-7,16-23; lanes 16-31 hold K 8-15,24-31).
// W is pre-swizzled the same way by k_cvt_w, so each lane's fragment is one
// contiguous 32-byte vector load.
__global__ __launch_bounds__(TPB, 1)
void k_mlp_layer(const __bf16* __restrict__ X, int M, int Kin, int Kpad,
                 const __bf16* __restrict__ W, int Nout,
                 const float* __restrict__ bias,
                 const float* __restrict__ gamma,
                 const float* __restrict__ beta,
                 int act, __bf16* __restrict__ Y) {
  __shared__ __align__(32) __bf16 sX[MAX_KTILES * 512];
  __shared__ float sY[16 * MAX_NOUT];

  const int tid  = threadIdx.x;
  const int wave = tid >> 5;
  const int lane = tid & 31;
  const int row0 = blockIdx.x * 16;
  const int ntiles = Nout >> 4;
  const int ktiles = Kpad >> 5;

  // ---- stage X tile into LDS in fragment order (no divisions) ----
  {
    const int ko    = lane;                        // k % 32 for this lane
    const int joff  = ((ko >> 4) << 3) + (ko & 7); // j within fragment
    const int lhalf = ((ko >> 3) & 1) << 4;        // +16 lane group
    for (int m = wave; m < 16; m += 8) {
      int gm = row0 + m;
      int dst = (m + lhalf) * 16 + joff;
      int kt = 0;
      for (int k = lane; k < Kpad; k += 32, ++kt) {
        float v = (gm < M && k < Kin) ? (float)X[gm * Kin + k] : 0.0f;
        sX[kt * 512 + dst] = (__bf16)v;
      }
    }
  }
  __syncthreads();

  // ---- WMMA: wave owns N tiles {wave, wave+8, ...}, processed in pairs ----
  const size_t wstep = (size_t)ntiles * 512;
  const int mrow = ((lane >> 4) << 3);  // C layout: m = r + 8*(lane>>4)
  const int ncol = lane & 15;           //           n = lane&15

  int nt = wave;
  for (; nt + 8 < ntiles; nt += 16) {   // dual-accumulator path (N >= 256)
    const __bf16* Wp0 = W + ((size_t)nt * 32 + lane) * 16;
    const __bf16* Wp1 = Wp0 + (size_t)8 * 512;
    v8f acc0 = {}, acc1 = {};
#pragma unroll 2
    for (int kt = 0; kt < ktiles; ++kt) {
      v16bf a  = *(const v16bf*)(sX + kt * 512 + lane * 16);
      v16bf b0 = *(const v16bf*)(Wp0 + (size_t)kt * wstep);
      v16bf b1 = *(const v16bf*)(Wp1 + (size_t)kt * wstep);
      __builtin_prefetch(Wp0 + (size_t)(kt + 1) * wstep, 0, 0);
      __builtin_prefetch(Wp1 + (size_t)(kt + 1) * wstep, 0, 0);
      acc0 = wmma_bf16(a, b0, acc0);
      acc1 = wmma_bf16(a, b1, acc1);
    }
    int n0 = nt * 16 + ncol, n1 = (nt + 8) * 16 + ncol;
    float bn0 = bias[n0], bn1 = bias[n1];
#pragma unroll
    for (int r = 0; r < 8; ++r) {
      int m = r + mrow;
      sY[m * Nout + n0] = acc0[r] + bn0;
      sY[m * Nout + n1] = acc1[r] + bn1;
    }
  }
  if (nt < ntiles) {                    // single-tile tail (N = 64/128)
    const __bf16* Wp = W + ((size_t)nt * 32 + lane) * 16;
    v8f acc = {};
#pragma unroll 2
    for (int kt = 0; kt < ktiles; ++kt) {
      v16bf a = *(const v16bf*)(sX + kt * 512 + lane * 16);
      v16bf b = *(const v16bf*)(Wp + (size_t)kt * wstep);
      __builtin_prefetch(Wp + (size_t)(kt + 1) * wstep, 0, 0);
      acc = wmma_bf16(a, b, acc);
    }
    int n = nt * 16 + ncol;
    float bn = bias[n];
#pragma unroll
    for (int r = 0; r < 8; ++r) sY[(r + mrow) * Nout + n] = acc[r] + bn;
  }
  __syncthreads();

  // ---- epilogue ----
  if (act == 1) {
    // LayerNorm over Nout + tanh. 16 threads per row, shfl reduce within a
    // 16-lane group (stays inside one wave32).
    int m = tid >> 4, j = tid & 15;
    float s = 0.0f;
    for (int c = j; c < Nout; c += 16) s += sY[m * Nout + c];
#pragma unroll
    for (int off = 8; off; off >>= 1) s += __shfl_xor(s, off, 16);
    float mu = s / (float)Nout;
    float vv = 0.0f;
    for (int c = j; c < Nout; c += 16) {
      float d = sY[m * Nout + c] - mu;
      vv += d * d;
    }
#pragma unroll
    for (int off = 8; off; off >>= 1) vv += __shfl_xor(vv, off, 16);
    float inv = rsqrtf(vv / (float)Nout + 1e-5f);
    int gm = row0 + m;
    if (gm < M) {
      for (int c = j; c < Nout; c += 16) {
        float yv = tanhf((sY[m * Nout + c] - mu) * inv * gamma[c] + beta[c]);
        Y[gm * Nout + c] = (__bf16)yv;
      }
    }
  } else {
    for (int m = wave; m < 16; m += 8) {
      int gm = row0 + m;
      if (gm < M)
        for (int c = lane; c < Nout; c += 32)
          Y[gm * Nout + c] = (__bf16)sY[m * Nout + c];
    }
  }
}

// ---------------- scalar sigmoid head (K -> 1), wave per row ---------------
__global__ __launch_bounds__(TPB)
void k_dot_sig(const __bf16* __restrict__ X, int M, int K,
               const float* __restrict__ W, const float* __restrict__ b,
               float* __restrict__ out) {
  int r = blockIdx.x * 8 + (threadIdx.x >> 5);
  int lane = threadIdx.x & 31;
  if (r >= M) return;
  float s = 0.0f;
  for (int k = lane; k < K; k += 32) s += (float)X[r * K + k] * W[k];
#pragma unroll
  for (int off = 16; off; off >>= 1) s += __shfl_xor(s, off, 32);
  if (lane == 0) out[r] = 1.0f / (1.0f + expf(-(s + b[0])));
}

// ---- weight fp32 -> bf16, swizzled into WMMA B-fragment order -------------
// dst layout: [kt][nt][lane][j], 16 contiguous bf16 per lane fragment.
__global__ void k_cvt_w(const float* __restrict__ src, __bf16* __restrict__ dst,
                        int K, int N, int Kpad, int ntiles) {
  int idx = blockIdx.x * blockDim.x + threadIdx.x;
  if (idx >= Kpad * N) return;
  int tile = idx >> 9;
  int rr   = idx & 511;
  int lane = rr >> 4;
  int j    = rr & 15;
  int kt = tile / ntiles;
  int nt = tile - kt * ntiles;
  int n  = nt * 16 + (lane & 15);
  int ko = (((lane >> 4) & 1) << 3) + ((j >> 3) << 4) + (j & 7);
  int k  = kt * 32 + ko;
  dst[idx] = (k < K) ? (__bf16)src[k * N + n] : (__bf16)0.0f;
}

__global__ void k_cvt_f2b(const float* __restrict__ s, __bf16* __restrict__ d,
                          int n) {
  int i = blockIdx.x * blockDim.x + threadIdx.x;
  if (i < n) d[i] = (__bf16)s[i];
}

__global__ void k_zero_f(float* __restrict__ p, int n) {
  int i = blockIdx.x * blockDim.x + threadIdx.x;
  if (i < n) p[i] = 0.0f;
}

// ---------------- concat/gather kernels ------------------------------------
__global__ void k_concat_eenc(const float* __restrict__ x,
                              const int* __restrict__ row,
                              const int* __restrict__ col,
                              const float* __restrict__ ea,
                              __bf16* __restrict__ out, int E) {
  int idx = blockIdx.x * blockDim.x + threadIdx.x;
  if (idx >= E * 10) return;
  int e = idx / 10, c = idx - e * 10;
  float v;
  if (c < 3)      v = x[row[e] * 3 + c];
  else if (c < 6) v = x[col[e] * 3 + (c - 3)];
  else            v = ea[e * 4 + (c - 6)];
  out[idx] = (__bf16)v;
}

__global__ void k_concat_initcls(const float* __restrict__ x,
                                 const int* __restrict__ row,
                                 const int* __restrict__ col,
                                 const float* __restrict__ ea,
                                 const __bf16* __restrict__ h,
                                 const __bf16* __restrict__ eh,
                                 __bf16* __restrict__ out, int E) {
  int idx = blockIdx.x * blockDim.x + threadIdx.x;
  if (idx >= E * 202) return;
  int e = idx / 202, c = idx - e * 202;
  float v;
  if (c < 64)       v = (float)h[row[e] * 64 + c];
  else if (c < 128) v = (float)h[col[e] * 64 + (c - 64)];
  else if (c < 192) v = (float)eh[e * 64 + (c - 128)];
  else if (c < 195) v = x[row[e] * 3 + (c - 192)];
  else if (c < 198) v = x[col[e] * 3 + (c - 195)];
  else              v = ea[e * 4 + (c - 198)];
  out[idx] = (__bf16)v;
}

__global__ void k_concat_ef(const float* __restrict__ x,
                            const int* __restrict__ row,
                            const int* __restrict__ col,
                            const float* __restrict__ ea,
                            const __bf16* __restrict__ h,
                            const __bf16* __restrict__ eh,
                            const float* __restrict__ lab,
                            __bf16* __restrict__ out, int E) {
  int idx = blockIdx.x * blockDim.x + threadIdx.x;
  if (idx >= E * 203) return;
  int e = idx / 203, c = idx - e * 203;
  float v;
  if (c < 64)        v = (float)h[row[e] * 64 + c];
  else if (c < 128)  v = (float)h[col[e] * 64 + (c - 64)];
  else if (c < 192)  v = (float)eh[e * 64 + (c - 128)];
  else if (c == 192) v = lab[e];
  else if (c < 196)  v = x[row[e] * 3 + (c - 193)];
  else if (c < 199)  v = x[col[e] * 3 + (c - 196)];
  else               v = ea[e * 4 + (c - 199)];
  out[idx] = (__bf16)v;
}

__global__ void k_concat_gat(const float* __restrict__ x,
                             const __bf16* __restrict__ h,
                             const float* __restrict__ one_in,
                             const float* __restrict__ one_out,
                             const float* __restrict__ two_in,
                             const float* __restrict__ two_out,
                             __bf16* __restrict__ out, int N) {
  int idx = blockIdx.x * blockDim.x + threadIdx.x;
  if (idx >= N * 579) return;
  int n = idx / 579, c = idx - n * 579;
  float v;
  if (c < 3)        v = x[n * 3 + c];
  else if (c < 67)  v = (float)h[n * 64 + (c - 3)];
  else if (c < 195) v = one_in[n * 128 + (c - 67)];
  else if (c < 323) v = one_out[n * 128 + (c - 195)];
  else if (c < 451) v = two_in[n * 128 + (c - 323)];
  else              v = two_out[n * 128 + (c - 451)];
  out[idx] = (__bf16)v;
}

// ---------------- segment-sum scatters (f32 atomics) -----------------------
__global__ void k_scatter1(const int* __restrict__ row,
                           const int* __restrict__ col,
                           const __bf16* __restrict__ h,
                           const __bf16* __restrict__ eh,
                           const float* __restrict__ w,
                           float* __restrict__ one_in,
                           float* __restrict__ one_out, int E) {
  int idx = blockIdx.x * blockDim.x + threadIdx.x;
  if (idx >= E * 128) return;
  int e = idx >> 7, c = idx & 127;
  float wt = w[e];
  int r = row[e], cl = col[e];
  float ehv = (c >= 64) ? (float)eh[e * 64 + (c - 64)] : 0.0f;
  float vin  = wt * ((c < 64) ? (float)h[r * 64 + c] : ehv);
  float vout = wt * ((c < 64) ? (float)h[cl * 64 + c] : ehv);
  atomicAdd(&one_in[cl * 128 + c], vin);
  atomicAdd(&one_out[r * 128 + c], vout);
}

__global__ void k_scatter2(const int* __restrict__ row,
                           const int* __restrict__ col,
                           const float* __restrict__ one_in,
                           const float* __restrict__ one_out,
                           const float* __restrict__ w,
                           float* __restrict__ two_in,
                           float* __restrict__ two_out, int E) {
  int idx = blockIdx.x * blockDim.x + threadIdx.x;
  if (idx >= E * 128) return;
  int e = idx >> 7, c = idx & 127;
  float wt = w[e];
  int r = row[e], cl = col[e];
  atomicAdd(&two_in[cl * 128 + c], wt * one_in[r * 128 + c]);
  atomicAdd(&two_out[r * 128 + c], wt * one_out[cl * 128 + c]);
}

// ===========================================================================
extern "C" void kernel_launch(void* const* d_in, const int* in_sizes, int n_in,
                              void* d_out, int out_size, void* d_ws,
                              size_t ws_size, hipStream_t stream) {
  (void)n_in; (void)out_size; (void)ws_size;
  const int N = in_sizes[0] / 3;   // 50000 nodes
  const int E = in_sizes[2] / 4;   // 500000 edges

  const float* x  = (const float*)d_in[0];
  const int*   ei = (const int*)d_in[1];
  const int*   row = ei;
  const int*   col = ei + E;
  const float* ea = (const float*)d_in[2];

  // JAX tree flatten order (sorted keys): edge_cls, edge_encoder, edge_mlp,
  // gat, init_cls, node_encoder; inside each MLP: Ws, betas, bs, gs.
  // edge_cls:     Ws 3-6,  betas 7-9,   bs 10-13, gs 14-16
  // edge_encoder: Ws 17-19,betas 20-21, bs 22-24, gs 25-26
  // edge_mlp:     Ws 27-29,betas 30-31, bs 32-34, gs 35-36
  // gat:          Ws 37-40,betas 41-43, bs 44-47, gs 48-50
  // init_cls:     Ws 51-53,betas 54-55, bs 56-58, gs 59-60
  // node_encoder: Ws 61-63,betas 64-65, bs 66-68, gs 69-70

  char* base = (char*)d_ws;
  size_t off = 0;
  auto alloc = [&](size_t bytes) -> void* {
    void* r = base + off;
    off = (off + bytes + 255) & ~(size_t)255;
    return r;
  };

  struct WSpec { int idx, K, Nn; };
  const WSpec wspec[18] = {
    {61, 3, 128},  {62, 128, 128}, {63, 128, 64},                // node_enc
    {17, 10, 128}, {18, 128, 128}, {19, 128, 64},                // edge_enc
    {51, 202, 256},{52, 256, 128},                               // init_cls
    {37, 579, 512},{38, 512, 256}, {39, 256, 128}, {40, 128, 64},// gat
    {27, 203, 256},{28, 256, 128}, {29, 128, 64},                // edge_mlp
    {3, 203, 256}, {4, 256, 128},  {5, 128, 64},                 // edge_cls
  };
  __bf16* wb[18];
  for (int i = 0; i < 18; ++i) {
    int Kpad = (wspec[i].K + 31) & ~31;
    wb[i] = (__bf16*)alloc((size_t)Kpad * wspec[i].Nn * sizeof(__bf16));
  }

  __bf16* h_bf  = (__bf16*)alloc((size_t)N * 64 * sizeof(__bf16));
  __bf16* eh_bf = (__bf16*)alloc((size_t)E * 64 * sizeof(__bf16));
  float*  elab  = (float*)alloc((size_t)E * sizeof(float));
  float*  agg   = (float*)alloc((size_t)N * 512 * sizeof(float));
  float* one_in  = agg;
  float* one_out = agg + (size_t)N * 128;
  float* two_in  = agg + (size_t)N * 256;
  float* two_out = agg + (size_t)N * 384;
  __bf16* actA = (__bf16*)alloc((size_t)E * 203 * sizeof(__bf16));
  __bf16* actB = (__bf16*)alloc((size_t)E * 256 * sizeof(__bf16));

  auto blocks = [](long long n) { return (unsigned)((n + TPB - 1) / TPB); };

  // ------------------- convert + swizzle weights to bf16 ----------------
  for (int i = 0; i < 18; ++i) {
    int K = wspec[i].K, Nn = wspec[i].Nn, Kpad = (K + 31) & ~31;
    k_cvt_w<<<blocks((long long)Kpad * Nn), TPB, 0, stream>>>(
        (const float*)d_in[wspec[i].idx], wb[i], K, Nn, Kpad, Nn >> 4);
  }

  auto layer = [&](const __bf16* X, int M, int Kin, const __bf16* W, int Nout,
                   int bi, int gi, int bei, int act, __bf16* Y) {
    int Kpad = (Kin + 31) & ~31;
    k_mlp_layer<<<(M + 15) / 16, TPB, 0, stream>>>(
        X, M, Kin, Kpad, W, Nout, (const float*)d_in[bi],
        gi >= 0 ? (const float*)d_in[gi] : nullptr,
        bei >= 0 ? (const float*)d_in[bei] : nullptr, act, Y);
  };

  // ------------------- node encoder: x -> h -----------------------------
  k_cvt_f2b<<<blocks((long long)N * 3), TPB, 0, stream>>>(x, actA, N * 3);
  layer(actA, N, 3,   wb[0], 128, 66, 69, 64, 1, actB);
  layer(actB, N, 128, wb[1], 128, 67, 70, 65, 1, actA);
  layer(actA, N, 128, wb[2], 64,  68, -1, -1, 0, h_bf);

  // ------------------- edge encoder -> edge_hidden ----------------------
  k_concat_eenc<<<blocks((long long)E * 10), TPB, 0, stream>>>(
      x, row, col, ea, actA, E);
  layer(actA, E, 10,  wb[3], 128, 22, 25, 20, 1, actB);
  layer(actB, E, 128, wb[4], 128, 23, 26, 21, 1, actA);
  layer(actA, E, 128, wb[5], 64,  24, -1, -1, 0, eh_bf);

  // ------------------- init_cls -> edge_labels --------------------------
  k_concat_initcls<<<blocks((long long)E * 202), TPB, 0, stream>>>(
      x, row, col, ea, h_bf, eh_bf, actA, E);
  layer(actA, E, 202, wb[6], 256, 56, 59, 54, 1, actB);
  layer(actB, E, 256, wb[7], 128, 57, 60, 55, 1, actA);
  k_dot_sig<<<(E + 7) / 8, TPB, 0, stream>>>(
      actA, E, 128, (const float*)d_in[53], (const float*)d_in[58], elab);

  // ------------------- message-passing iterations -----------------------
  for (int it = 0; it < 6; ++it) {
    // GAT aggregation (2 hops, in/out)
    k_zero_f<<<blocks((long long)N * 512), TPB, 0, stream>>>(agg, N * 512);
    k_scatter1<<<blocks((long long)E * 128), TPB, 0, stream>>>(
        row, col, h_bf, eh_bf, elab, one_in, one_out, E);
    k_scatter2<<<blocks((long long)E * 128), TPB, 0, stream>>>(
        row, col, one_in, one_out, elab, two_in, two_out, E);
    k_concat_gat<<<blocks((long long)N * 579), TPB, 0, stream>>>(
        x, h_bf, one_in, one_out, two_in, two_out, actA, N);
    layer(actA, N, 579, wb[8],  512, 44, 48, 41, 1, actB);
    layer(actB, N, 512, wb[9],  256, 45, 49, 42, 1, actA);
    layer(actA, N, 256, wb[10], 128, 46, 50, 43, 1, actB);
    layer(actB, N, 128, wb[11], 64,  47, -1, -1, 0, h_bf);

    // edge_mlp -> new edge_hidden
    k_concat_ef<<<blocks((long long)E * 203), TPB, 0, stream>>>(
        x, row, col, ea, h_bf, eh_bf, elab, actA, E);
    layer(actA, E, 203, wb[12], 256, 32, 35, 30, 1, actB);
    layer(actB, E, 256, wb[13], 128, 33, 36, 31, 1, actA);
    layer(actA, E, 128, wb[14], 64,  34, -1, -1, 0, eh_bf);

    // edge_cls -> new edge_labels
    k_concat_ef<<<blocks((long long)E * 203), TPB, 0, stream>>>(
        x, row, col, ea, h_bf, eh_bf, elab, actA, E);
    layer(actA, E, 203, wb[15], 256, 10, 14, 7, 1, actB);
    layer(actB, E, 256, wb[16], 128, 11, 15, 8, 1, actA);
    layer(actA, E, 128, wb[17], 64,  12, 16, 9, 1, actB);
    k_dot_sig<<<(E + 7) / 8, TPB, 0, stream>>>(
        actB, E, 64, (const float*)d_in[6], (const float*)d_in[13], elab);
  }

  hipMemcpyAsync(d_out, elab, (size_t)E * sizeof(float),
                 hipMemcpyDeviceToDevice, stream);
}
